// AttentionBlock_88716844466872
// MI455X (gfx1250) — matrix operand order repaired and checked by reading, not compiled, not run on hardware
//
#include <hip/hip_runtime.h>

typedef unsigned short u16;
typedef unsigned int   u32;
typedef unsigned long long u64;

typedef __attribute__((ext_vector_type(16))) __bf16 bf16x16;
typedef __attribute__((ext_vector_type(8)))  float  f32x8;

constexpr int CH   = 512;   // channels
constexpr int HWN  = 4096;  // H*W
constexpr int BATCH = 4;

union FragAB { bf16x16 v; u32 u[8]; };

__device__ __forceinline__ u16 f2bf(float x) {
  u32 u = __builtin_bit_cast(u32, x);
  u32 r = (u + 0x7FFFu + ((u >> 16) & 1u)) >> 16;
  return (u16)r;
}

// ---- CDNA5 async global->LDS DMA (ASYNCcnt-tracked), ISA 10.7 / 15.18.3 ----
__device__ __forceinline__ void async_copy_b128(u16* lds_dst, const u16* gsrc) {
  u32 l = (u32)(size_t)lds_dst;          // generic LDS addr[31:0] == LDS offset
  u64 g = (u64)(size_t)gsrc;
  asm volatile("global_load_async_to_lds_b128 %0, %1, off"
               :: "v"(l), "v"(g) : "memory");
}
__device__ __forceinline__ void wait_async() {
  asm volatile("s_wait_asynccnt 0x0" ::: "memory");
}

// A-matrix 16x32 bf16 fragment (ISA 7.12.2): lane M = L&15, half = L>>4,
// VGPR v holds K pair: Kp = (v<4 ? 2v : 16+2(v-4)) + 8*half. tile is [M][K], ld even.
__device__ __forceinline__ bf16x16 load_frag_A(const u16* tile, int ld, int lane) {
  FragAB f;
  const int M = lane & 15, half = lane >> 4;
#pragma unroll
  for (int v = 0; v < 8; ++v) {
    const int kp = ((v < 4) ? (2 * v) : (16 + 2 * (v - 4))) + 8 * half;
    f.u[v] = *(const u32*)(tile + M * ld + kp);
  }
  return f.v;
}

// B-matrix 32x16 bf16 fragment: lane N = L&15, half = L>>4, VGPR v holds K pair
// Kp = (v<4 ? 2v : 16+2(v-4)) + 8*half. tile is [N][K], ld even.
__device__ __forceinline__ bf16x16 load_frag_B(const u16* tile, int ld, int lane) {
  FragAB f;
  const int N = lane & 15, kb = (lane >> 4) * 8;
#pragma unroll
  for (int v = 0; v < 8; ++v)
    f.u[v] = *(const u32*)(tile + N * ld + kb + ((v < 4) ? (2 * v) : (16 + 2 * (v - 4))));
  return f.v;
}

__device__ __forceinline__ float redmax16(float v) {
#pragma unroll
  for (int d = 1; d < 16; d <<= 1) v = fmaxf(v, __shfl_xor(v, d, 16));
  return v;
}
__device__ __forceinline__ float redsum16(float v) {
#pragma unroll
  for (int d = 1; d < 16; d <<= 1) v += __shfl_xor(v, d, 16);
  return v;
}

// ---------------- Kernel 0: weights fp32 (j,c) -> bf16 transposed (c,j) ----------------
__global__ __launch_bounds__(256)
void prep_w_kernel(const float* __restrict__ wq, const float* __restrict__ wk,
                   const float* __restrict__ wv, const float* __restrict__ wo,
                   u16* __restrict__ wT) {
  const int w = blockIdx.y;
  const float* src = (w == 0) ? wq : (w == 1) ? wk : (w == 2) ? wv : wo;
  u16* dst = wT + (size_t)w * CH * CH;
  const int i = blockIdx.x * 256 + threadIdx.x;   // dst index: c*512 + j
  const int c = i >> 9, j = i & 511;
  dst[i] = f2bf(src[j * CH + c]);
}

// ---------------- Kernel 1: GroupNorm -> bf16 (keeps x's flat layout) ----------------
__global__ __launch_bounds__(256)
void gn_kernel(const float* __restrict__ x, const float* __restrict__ gw,
               const float* __restrict__ gb, u16* __restrict__ hb) {
  const int b = blockIdx.x >> 5, g = blockIdx.x & 31;
  const size_t base = ((size_t)b * CH + (size_t)g * 16) * HWN;   // 16 ch/group * 4096
  const float4* xv = (const float4*)(x + base);
  const int t = threadIdx.x;

  float s = 0.f, ss = 0.f;
  for (int i = t; i < 16384; i += 256) {
    float4 v = xv[i];
    s  += v.x + v.y + v.z + v.w;
    ss += v.x * v.x + v.y * v.y + v.z * v.z + v.w * v.w;
  }
  __shared__ float rs[256], rq[256];
  rs[t] = s; rq[t] = ss;
  __syncthreads();
  for (int d = 128; d > 0; d >>= 1) {
    if (t < d) { rs[t] += rs[t + d]; rq[t] += rq[t + d]; }
    __syncthreads();
  }
  const float mean = rs[0] * (1.f / 65536.f);
  const float var  = rq[0] * (1.f / 65536.f) - mean * mean;
  const float rstd = rsqrtf(var + 1e-5f);

  u16* hv = hb + base;
  for (int i = t; i < 16384; i += 256) {
    const int cl = (i * 4) >> 12;           // channel within group
    const int c  = g * 16 + cl;
    const float sc = gw[c] * rstd;
    const float sh = gb[c] - mean * sc;
    float4 v = xv[i];
    u64 p = (u64)f2bf(v.x * sc + sh)
          | ((u64)f2bf(v.y * sc + sh) << 16)
          | ((u64)f2bf(v.z * sc + sh) << 32)
          | ((u64)f2bf(v.w * sc + sh) << 48);
    *(u64*)(hv + (size_t)i * 4) = p;
  }
}

// ---------------- Shared tiled-GEMM core: 128x128 tile, K-step 32 ----------------
// A: row-major (4096,512) bf16 at Arow (already offset to row p0), lda = 512.
// B: row-major (512,512) bf16, [k][n].
__device__ __forceinline__ void gemm_core(const u16* __restrict__ Arow,
                                          const u16* __restrict__ Bmat,
                                          int n0, u16* lA, u16* lB, f32x8 acc[4][2]) {
  const int t = threadIdx.x, lane = t & 31;
  const int wm = (t >> 5) >> 2, wn = (t >> 5) & 3;   // waves 2(M) x 4(N)
#pragma unroll 1
  for (int ks = 0; ks < 16; ++ks) {
    const int kc = ks * 32;
    __syncthreads();
    { // stage A tile 128x32 (row-major, stride 40) via async DMA to LDS
      const int m = t >> 1, hf = t & 1;
      const u16* g = Arow + (size_t)m * CH + kc + hf * 16;
      u16* l = lA + m * 40 + hf * 16;
      async_copy_b128(l, g);
      async_copy_b128(l + 8, g + 8);
    }
    { // stage B tile transposed to [n][k] (stride 40) -- scalar (reshuffles data)
      const int c = t >> 3, jg = t & 7;
      const uint4* src = (const uint4*)(Bmat + (size_t)(kc + c) * CH + n0 + jg * 16);
      uint4 d0 = src[0], d1 = src[1];
      if (ks < 15) __builtin_prefetch(Bmat + (size_t)(kc + 32 + c) * CH + n0 + jg * 16, 0, 1);
      u32 a8[8] = {d0.x, d0.y, d0.z, d0.w, d1.x, d1.y, d1.z, d1.w};
#pragma unroll
      for (int i = 0; i < 8; ++i) {
        lB[(jg * 16 + 2 * i    ) * 40 + c] = (u16)a8[i];
        lB[(jg * 16 + 2 * i + 1) * 40 + c] = (u16)(a8[i] >> 16);
      }
    }
    wait_async();
    __syncthreads();
    bf16x16 af[4], bfr[2];
#pragma unroll
    for (int mf = 0; mf < 4; ++mf) af[mf] = load_frag_A(lA + (wm * 64 + mf * 16) * 40, 40, lane);
#pragma unroll
    for (int nf = 0; nf < 2; ++nf) bfr[nf] = load_frag_B(lB + (wn * 32 + nf * 16) * 40, 40, lane);
#pragma unroll
    for (int mf = 0; mf < 4; ++mf)
#pragma unroll
      for (int nf = 0; nf < 2; ++nf)
        acc[mf][nf] = __builtin_amdgcn_wmma_f32_16x16x32_bf16(
            false, af[mf], false, bfr[nf], (short)0, acc[mf][nf], false, false);
  }
}

// ---------------- Kernel 2: QKV projections ----------------
__global__ __launch_bounds__(256)
void qkv_kernel(const u16* __restrict__ hb, const u16* __restrict__ wT,
                const float* __restrict__ bq, const float* __restrict__ bk,
                const float* __restrict__ bv,
                u16* __restrict__ qb, u16* __restrict__ kb, u16* __restrict__ vb) {
  __shared__ u16 lA[128 * 40], lB[128 * 40];
  const int b   = blockIdx.z;
  const int sel = blockIdx.y >> 2;
  const int n0  = (blockIdx.y & 3) * 128;
  const int p0  = blockIdx.x * 128;
  const u16* Bm = wT + (size_t)sel * CH * CH;
  const float* bias = sel == 0 ? bq : sel == 1 ? bk : bv;
  u16* out = (sel == 0 ? qb : sel == 1 ? kb : vb) + (size_t)b * CH * HWN;
  const u16* Arow = hb + (size_t)b * CH * HWN + (size_t)p0 * CH;

  const f32x8 zero8 = {0.f, 0.f, 0.f, 0.f, 0.f, 0.f, 0.f, 0.f};
  f32x8 acc[4][2];
#pragma unroll
  for (int mf = 0; mf < 4; ++mf)
#pragma unroll
    for (int nf = 0; nf < 2; ++nf) acc[mf][nf] = zero8;

  gemm_core(Arow, Bm, n0, lA, lB, acc);

  const int lane = threadIdx.x & 31;
  const int wm = (threadIdx.x >> 5) >> 2, wn = (threadIdx.x >> 5) & 3;
  const int half = lane >> 4, l15 = lane & 15;
#pragma unroll
  for (int nf = 0; nf < 2; ++nf) {
    const int n = n0 + wn * 32 + nf * 16 + l15;
    const float bb = bias[n];
#pragma unroll
    for (int mf = 0; mf < 4; ++mf)
#pragma unroll
      for (int v = 0; v < 8; ++v) {
        const int p = p0 + wm * 64 + mf * 16 + half * 8 + v;
        out[(size_t)p * CH + n] = f2bf(acc[mf][nf][v] + bb);
      }
  }
}

// ---------------- Kernel 3: single-pass flash attention ----------------
// q2/k2/v2 are the flat (C,HW) reinterpretations. Produces ot[n][c] (HW,C) bf16,
// already normalized by the softmax denominator.
__global__ __launch_bounds__(256)
void attn_flash_kernel(const u16* __restrict__ qbuf, const u16* __restrict__ kbuf,
                       const u16* __restrict__ vbuf, u16* __restrict__ obuf) {
  const int b  = blockIdx.y;
  const int n0 = blockIdx.x * 32;
  const size_t boff = (size_t)b * CH * HWN;
  const u16* q2 = qbuf + boff;
  const u16* k2 = kbuf + boff;
  const u16* v2 = vbuf + boff;

  __shared__ u16  lQ[32 * 40];       // q^T tile [n][c], restaged per k-step
  __shared__ u16  lKV[512 * 40];     // phase1: K tile [m(128)][c(32)]; phase4: V chunk [c(512)][m(32)]
  __shared__ u16  lP[32 * 132];      // P tile [n][m(128)] bf16
  __shared__ float lRed[8 * 32];
  __shared__ float lM[32], lL[32], lF[32];

  const int t = threadIdx.x, lane = t & 31, w = t >> 5;
  const int half = lane >> 4, l15 = lane & 15;

  if (t < 32) { lM[t] = -3.0e38f; lL[t] = 0.f; }

  const f32x8 zero8 = {0.f, 0.f, 0.f, 0.f, 0.f, 0.f, 0.f, 0.f};
  f32x8 O[2][4];                       // wave owns c in [w*64, w*64+64)
#pragma unroll
  for (int nf = 0; nf < 2; ++nf)
#pragma unroll
    for (int cf = 0; cf < 4; ++cf) O[nf][cf] = zero8;

#pragma unroll 1
  for (int m0 = 0; m0 < HWN; m0 += 128) {
    // ---- Phase 1: S(32n x 128m) = q^T k ; waves split the 128 m-columns (16 each)
    f32x8 S[2] = {zero8, zero8};
#pragma unroll 1
    for (int ks = 0; ks < 16; ++ks) {
      const int kc = ks * 32;
      __syncthreads();
      { // stage lQ [n(32)][c(32)] from q2[c][n] (transposing)
        const int cidx = t >> 3, ng = t & 7;
        u64 d = *(const u64*)(q2 + (size_t)(kc + cidx) * HWN + n0 + ng * 4);
#pragma unroll
        for (int i = 0; i < 4; ++i) lQ[(ng * 4 + i) * 40 + cidx] = (u16)(d >> (16 * i));
      }
      { // stage lK [m(128)][c(32)] from k2[c][m] (transposing)
        const int cp = t >> 4, mg = t & 15;
#pragma unroll
        for (int cc = 0; cc < 2; ++cc) {
          const int c = cp * 2 + cc;
          const uint4 d = *(const uint4*)(k2 + (size_t)(kc + c) * HWN + m0 + mg * 8);
          u32 a4[4] = {d.x, d.y, d.z, d.w};
#pragma unroll
          for (int i = 0; i < 4; ++i) {
            lKV[(mg * 8 + 2 * i    ) * 40 + c] = (u16)a4[i];
            lKV[(mg * 8 + 2 * i + 1) * 40 + c] = (u16)(a4[i] >> 16);
          }
        }
      }
      __syncthreads();
      bf16x16 bk = load_frag_B(lKV + w * 16 * 40, 40, lane);
#pragma unroll
      for (int nf = 0; nf < 2; ++nf) {
        bf16x16 aq = load_frag_A(lQ + nf * 16 * 40, 40, lane);
        S[nf] = __builtin_amdgcn_wmma_f32_16x16x32_bf16(
            false, aq, false, bk, (short)0, S[nf], false, false);
      }
    }

    // ---- Phase 2: online softmax stats
#pragma unroll
    for (int nf = 0; nf < 2; ++nf)
#pragma unroll
      for (int v = 0; v < 8; ++v) {
        float mx = redmax16(S[nf][v]);
        if (l15 == 0) lRed[w * 32 + nf * 16 + half * 8 + v] = mx;
      }
    __syncthreads();
    if (t < 32) {
      float mo = lM[t], mw = mo;
#pragma unroll
      for (int i = 0; i < 8; ++i) mw = fmaxf(mw, lRed[i * 32 + t]);
      lF[t] = __expf(mo - mw);
      lM[t] = mw;
    }
    __syncthreads();
#pragma unroll
    for (int nf = 0; nf < 2; ++nf)
#pragma unroll
      for (int v = 0; v < 8; ++v) {
        const int row = nf * 16 + half * 8 + v;
        const float p = __expf(S[nf][v] - lM[row]);
        lP[row * 132 + w * 16 + l15] = f2bf(p);
        float rs = redsum16(p);
        if (l15 == 0) lRed[w * 32 + row] = rs;
      }
    __syncthreads();
    if (t < 32) {
      float l = lL[t] * lF[t];
#pragma unroll
      for (int i = 0; i < 8; ++i) l += lRed[i * 32 + t];
      lL[t] = l;
    }
    // ---- Phase 3: rescale O accumulators
#pragma unroll
    for (int nf = 0; nf < 2; ++nf)
#pragma unroll
      for (int v = 0; v < 8; ++v) {
        const float f = lF[nf * 16 + half * 8 + v];
#pragma unroll
        for (int cf = 0; cf < 4; ++cf) O[nf][cf][v] *= f;
      }

    // ---- Phase 4: O += P(32n x 32m) * V^T(32m x 512c), m chunked by 32
#pragma unroll 1
    for (int ms = 0; ms < 128; ms += 32) {
      __syncthreads();
      { // stage V chunk [c(512)][m(32)] via async DMA to LDS (layout-preserving)
        const int c0 = t * 2;
#pragma unroll
        for (int cc = 0; cc < 2; ++cc) {
          const int c = c0 + cc;
          const u16* g = v2 + (size_t)c * HWN + m0 + ms;
          u16* l = lKV + c * 40;
#pragma unroll
          for (int i = 0; i < 4; ++i)
            async_copy_b128(l + i * 8, g + i * 8);
        }
        wait_async();
      }
      __syncthreads();
#pragma unroll
      for (int nf = 0; nf < 2; ++nf) {
        bf16x16 ap = load_frag_A(lP + nf * 16 * 132 + ms, 132, lane);
#pragma unroll
        for (int cf = 0; cf < 4; ++cf) {
          bf16x16 bvf = load_frag_B(lKV + (w * 64 + cf * 16) * 40, 40, lane);
          O[nf][cf] = __builtin_amdgcn_wmma_f32_16x16x32_bf16(
              false, ap, false, bvf, (short)0, O[nf][cf], false, false);
        }
      }
    }
  }

  __syncthreads();
  // ---- Epilogue: divide by softmax denominator, store ot (HW, C) bf16
#pragma unroll
  for (int nf = 0; nf < 2; ++nf)
#pragma unroll
    for (int v = 0; v < 8; ++v) {
      const int row = nf * 16 + half * 8 + v;
      const float inv = 1.0f / lL[row];
#pragma unroll
      for (int cf = 0; cf < 4; ++cf) {
        const int c = w * 64 + cf * 16 + l15;
        obuf[boff + (size_t)(n0 + row) * CH + c] = f2bf(O[nf][cf][v] * inv);
      }
    }
}

// ---------------- Kernel 4: out-projection + bias + residual (fp32 out) ----------------
__global__ __launch_bounds__(256)
void outproj_kernel(const u16* __restrict__ ot, const u16* __restrict__ woT,
                    const float* __restrict__ bo, const float* __restrict__ x,
                    float* __restrict__ y) {
  __shared__ u16 lA[128 * 40], lB[128 * 40];
  const int b  = blockIdx.z;
  const int n0 = blockIdx.y * 128;
  const int p0 = blockIdx.x * 128;
  const size_t boff = (size_t)b * CH * HWN;
  const u16* Arow = ot + boff + (size_t)p0 * CH;

  const f32x8 zero8 = {0.f, 0.f, 0.f, 0.f, 0.f, 0.f, 0.f, 0.f};
  f32x8 acc[4][2];
#pragma unroll
  for (int mf = 0; mf < 4; ++mf)
#pragma unroll
    for (int nf = 0; nf < 2; ++nf) acc[mf][nf] = zero8;

  gemm_core(Arow, woT, n0, lA, lB, acc);

  const int lane = threadIdx.x & 31;
  const int wm = (threadIdx.x >> 5) >> 2, wn = (threadIdx.x >> 5) & 3;
  const int half = lane >> 4, l15 = lane & 15;
#pragma unroll
  for (int nf = 0; nf < 2; ++nf) {
    const int n = n0 + wn * 32 + nf * 16 + l15;
    const float bb = bo[n];
#pragma unroll
    for (int mf = 0; mf < 4; ++mf)
#pragma unroll
      for (int v = 0; v < 8; ++v) {
        const int p = p0 + wm * 64 + mf * 16 + half * 8 + v;
        const size_t idx = boff + (size_t)p * CH + n;
        y[idx] = acc[mf][nf][v] + bb + x[idx];
      }
  }
}

// ---------------- Launcher ----------------
extern "C" void kernel_launch(void* const* d_in, const int* in_sizes, int n_in,
                              void* d_out, int out_size, void* d_ws, size_t ws_size,
                              hipStream_t stream) {
  (void)in_sizes; (void)n_in; (void)out_size; (void)ws_size;
  const float* x  = (const float*)d_in[0];
  const float* gw = (const float*)d_in[1];
  const float* gb = (const float*)d_in[2];
  const float* wq = (const float*)d_in[3];
  const float* bq = (const float*)d_in[4];
  const float* wk = (const float*)d_in[5];
  const float* bk = (const float*)d_in[6];
  const float* wv = (const float*)d_in[7];
  const float* bv = (const float*)d_in[8];
  const float* wo = (const float*)d_in[9];
  const float* bo = (const float*)d_in[10];
  float* y = (float*)d_out;

  char* ws = (char*)d_ws;
  const size_t tb = (size_t)BATCH * CH * HWN * sizeof(u16);  // 16 MB per bf16 tensor
  u16* hb = (u16*)(ws);
  u16* qb = (u16*)(ws + tb);
  u16* kb = (u16*)(ws + 2 * tb);
  u16* vb = (u16*)(ws + 3 * tb);
  u16* ot = (u16*)(ws + 4 * tb);
  u16* wT = (u16*)(ws + 5 * tb);                             // 4 x 512KB bf16 weights

  prep_w_kernel<<<dim3(1024, 4), 256, 0, stream>>>(wq, wk, wv, wo, wT);
  gn_kernel<<<dim3(128), 256, 0, stream>>>(x, gw, gb, hb);
  qkv_kernel<<<dim3(32, 12, 4), 256, 0, stream>>>(hb, wT, bq, bk, bv, qb, kb, vb);
  attn_flash_kernel<<<dim3(128, 4), 256, 0, stream>>>(qb, kb, vb, ot);
  outproj_kernel<<<dim3(32, 4, 4), 256, 0, stream>>>(ot, wT + 3 * (size_t)CH * CH, bo, x, y);
}
